// TRFlow_53480932770042
// MI455X (gfx1250) — matrix operand order repaired and checked by reading, not compiled, not run on hardware
//
#include <hip/hip_runtime.h>
#include <hip/hip_bf16.h>

// ---------------------------------------------------------------------------
// CDNA5 (gfx1250) implementation of the TRFlow (PWC-style) optical flow net.
// Conv 3x3 layers run as implicit-GEMM on v_wmma_f32_16x16x32_bf16 (wave32).
// Weight tiles are staged to LDS by the Tensor Data Mover (tensor_load_to_lds,
// TENSORcnt-synchronized, HW OOB-zeroing + LDS row padding); activations are
// im2col-gathered to LDS as bf16. Dense-block concats are zero-copy via
// channel-offset strided NCHW buffers. Warp / correlation / resize are VALU.
// ---------------------------------------------------------------------------

typedef __bf16 bf16_t;
typedef bf16_t v16bf __attribute__((ext_vector_type(16)));
typedef float  v8f   __attribute__((ext_vector_type(8)));
typedef unsigned int v4u __attribute__((ext_vector_type(4)));
typedef int v8i_t __attribute__((ext_vector_type(8)));
typedef int v4i_t __attribute__((ext_vector_type(4)));

union BF16x16 { v16bf v; bf16_t e[16]; };
union F32x8   { v8f   v; float  e[8];  };

__device__ __forceinline__ bf16_t f2bf(float f) {
  union { float f; unsigned u; } in; in.f = f;
  unsigned r = in.u + 0x7FFFu + ((in.u >> 16) & 1u);   // RNE truncate to bf16
  union { unsigned short s; bf16_t b; } out;
  out.s = (unsigned short)(r >> 16);
  return out.b;
}

__device__ __forceinline__ float leaky(float v) { return v >= 0.f ? v : 0.1f * v; }

__device__ __forceinline__ int rfl(unsigned v) {
  return __builtin_amdgcn_readfirstlane((int)v);
}

#define LDSS 34   // A-tile LDS row stride in halves
#define BROW 33   // B-tile LDS row stride in dwords (32 data + 1 TDM pad)

// ---------------------------------------------------------------------------
// Implicit-GEMM conv3x3 ("same" padding = dil), bf16 WMMA, f32 accum.
// GEMM view per batch image: M = Hout*Wout, N = Cout, K = Cin*9.
// Block = 128 threads = 4 waves; block tile = 32 pixels x 64 couts.
// Wave wv owns couts [blockIdx.y*64 + wv*16, +16) and both 16-pixel A tiles.
// Weight K-slab (16 couts x 32 K, f32, row stride K) is DMA'd by the TDM
// into LDS with 1-dword row padding; OOB rows/cols are zero-filled by HW.
// ---------------------------------------------------------------------------
__global__ __launch_bounds__(128) void k_conv3x3_wmma(
    const float* __restrict__ x, int CinTot, int cin_off, int Hin, int Win,
    const float* __restrict__ w, const float* __restrict__ bias,
    float* __restrict__ y, int CoutTot, int cout_off, int Hout, int Wout,
    int Cin, int Cout, int stride, int dil, int relu)
{
  __shared__ bf16_t lA[32 * LDSS];       // two 16x32 bf16 A tiles
  __shared__ float  lB[4][16 * BROW];    // per-wave 16x32 f32 B tile (padded)

  const int tid   = threadIdx.x;
  const int lane  = tid & 31;
  const int wv    = tid >> 5;
  const int b     = blockIdx.z;
  const int m0    = blockIdx.x * 32;
  const int ntile = blockIdx.y * 64 + wv * 16;
  const int M     = Hout * Wout;
  const int K     = Cin * 9;
  const long long HWin = (long long)Hin * Win;
  const float* xb = x + ((long long)b * CinTot + cin_off) * HWin;

  // ---- loop-invariant A-staging coordinates (per thread) ------------------
  const int am  = tid >> 2;          // pixel row in block tile: 0..31
  const int akb = (tid & 3) * 8;     // K sub-slab base: 0,8,16,24
  const int apm = m0 + am;
  const bool am_ok = apm < M;
  const int aoh = apm / Wout;
  const int aow = apm - aoh * Wout;
  const int ihb = aoh * stride - dil;
  const int iwb = aow * stride - dil;

  // ---- TDM destination (uniform per wave) ---------------------------------
  const unsigned ldsB = (unsigned)(uintptr_t)&lB[wv][0];  // flat[31:0] == LDS offset

  F32x8 acc0, acc1;
#pragma unroll
  for (int r = 0; r < 8; ++r) { acc0.e[r] = 0.f; acc1.e[r] = 0.f; }

  for (int k0 = 0; k0 < K; k0 += 32) {
    __syncthreads();   // waits own DScnt, then barrier: lA safe to overwrite

    // ---- TDM: async-load weight slab (16 couts x 32 K, f32) into LDS ------
    {
      const unsigned long long ga =
          (unsigned long long)(uintptr_t)(w + (size_t)ntile * K + k0);
      const unsigned td0 = (unsigned)(K - k0);                         // valid K
      const unsigned td1 = (Cout > ntile) ? (unsigned)(Cout - ntile) : 0u;
      v4u g0;
      g0.x = (unsigned)rfl(1u);                         // count=1, user desc
      g0.y = (unsigned)rfl(ldsB);                       // lds_addr (bytes)
      g0.z = (unsigned)rfl((unsigned)ga);               // global_addr[31:0]
      g0.w = (unsigned)rfl((unsigned)((ga >> 32) & 0x1FFFFFFull) | (2u << 30));
      v8i_t g1;
      g1[0] = rfl((2u << 16) | (1u << 20) | (4u << 22));  // 4B elems, pad 1dw/32dw
      g1[1] = rfl((td0 & 0xFFFFu) << 16);                 // tensor_dim0 lo
      g1[2] = rfl((td0 >> 16) | ((td1 & 0xFFFFu) << 16)); // dim0 hi | dim1 lo
      g1[3] = rfl((td1 >> 16) | (32u << 16));             // dim1 hi | tile_dim0=32
      g1[4] = rfl(16u);                                   // tile_dim1=16
      g1[5] = rfl((unsigned)K);                           // dim0_stride lo (elems)
      g1[6] = rfl(0u);
      g1[7] = rfl(0u);
      v4i_t gz4 = {0, 0, 0, 0};
      v8i_t gz8 = {0, 0, 0, 0, 0, 0, 0, 0};
      __builtin_amdgcn_tensor_load_to_lds(g0, g1, gz4, gz4, gz8, 0);
    }

    // ---- stage A tile: 32 pixels x 32 K, bf16, incremental (c,ky,kx) ------
    {
      int gk = k0 + akb;
      int c  = gk / 9;
      int t  = gk - 9 * c;
      int ky = t / 3;
      int kx = t - 3 * ky;
#pragma unroll
      for (int j = 0; j < 8; ++j) {
        float v = 0.f;
        if (am_ok && gk < K) {
          const int ih = ihb + ky * dil;
          const int iw = iwb + kx * dil;
          if (ih >= 0 && ih < Hin && iw >= 0 && iw < Win)
            v = xb[(long long)c * HWin + (long long)ih * Win + iw];
        }
        lA[am * LDSS + akb + j] = f2bf(v);
        ++gk; ++kx;
        if (kx == 3) { kx = 0; ++ky; if (ky == 3) { ky = 0; ++c; } }
      }
    }
    __syncthreads();                        // A visible to all waves
    __builtin_amdgcn_s_wait_tensorcnt(0);   // B slab resident in LDS

    // ---- build fragments (wave32 WMMA layouts, ISA 7.12.2) ----------------
    const int hi   = lane >> 4;   // 0: lanes 0-15, 1: lanes 16-31
    const int mrow = lane & 15;
    BF16x16 a0, a1, bf;
#pragma unroll
    for (int h = 0; h < 16; ++h) {
      // A 16x32: lanes<16 hold K 0..15; lanes>=16 hold K 16..31
      const int ka = 16 * hi + h;
      a0.e[h] = lA[mrow * LDSS + ka];
      a1.e[h] = lA[(mrow + 16) * LDSS + ka];
    }
#pragma unroll
    for (int h = 0; h < 16; ++h) {
      // B 32x16: N = lane&15; lanes<16 K=0..15, lanes>=16 K=16..31
      bf.e[h] = f2bf(lB[wv][mrow * BROW + 16 * hi + h]);
    }
    acc0.v = __builtin_amdgcn_wmma_f32_16x16x32_bf16(
        false, a0.v, false, bf.v, (short)0, acc0.v, false, false);
    acc1.v = __builtin_amdgcn_wmma_f32_16x16x32_bf16(
        false, a1.v, false, bf.v, (short)0, acc1.v, false, false);
  }

  // ---- store C: VGPR r -> M = r + 8*(lane>=16), N = lane&15 ---------------
  const int col = ntile + (lane & 15);
  const int mh  = (lane >> 4) * 8;
  if (col < Cout) {
    const float bv = bias[col];
    float* yb = y + ((long long)b * CoutTot + cout_off + col) * M;
#pragma unroll
    for (int r = 0; r < 8; ++r) {
      const int pm0 = m0 + r + mh;
      if (pm0 < M) {
        float v = acc0.e[r] + bv;
        if (relu) v = leaky(v);
        yb[pm0] = v;
      }
      const int pm1 = m0 + 16 + r + mh;
      if (pm1 < M) {
        float v = acc1.e[r] + bv;
        if (relu) v = leaky(v);
        yb[pm1] = v;
      }
    }
  }
}

// ---------------------------------------------------------------------------
// Correlation: out[b, dy*9+dx, p] = leaky(mean_c x1[b,c,p] * x2w[b,c,p+d])
// ---------------------------------------------------------------------------
__global__ void k_corr(const float* __restrict__ x1, const float* __restrict__ x2w,
                       float* __restrict__ out, int CoTot, int coffo,
                       int C, int H, int W, int B)
{
  const int HW  = H * W;
  const int tot = B * 81 * HW;
  const int i   = blockIdx.x * blockDim.x + threadIdx.x;
  if (i >= tot) return;
  const int p = i % HW;
  const int o = (i / HW) % 81;
  const int b = i / (81 * HW);
  const int h = p / W, wp = p - (p / W) * W;
  const int dy = o / 9 - 4, dx = o % 9 - 4;
  const int h2 = h + dy, w2 = wp + dx;
  float s = 0.f;
  if (h2 >= 0 && h2 < H && w2 >= 0 && w2 < W) {
    const float* a  = x1  + (long long)b * C * HW + p;
    const float* bb = x2w + (long long)b * C * HW + (long long)h2 * W + w2;
    for (int c = 0; c < C; ++c)
      s += a[(long long)c * HW] * bb[(long long)c * HW];
  }
  s *= (1.f / (float)C);
  out[((long long)b * CoTot + coffo + o) * HW + p] = leaky(s);
}

// ---------------------------------------------------------------------------
// Flow warp: bilinear gather with out-of-range masking (mask>=1 rule).
// ---------------------------------------------------------------------------
__global__ void k_warp(const float* __restrict__ x, const float* __restrict__ flow,
                       float* __restrict__ out, int C, int H, int W, int B,
                       float sx, float sy)
{
  const int HW  = H * W;
  const int tot = B * C * HW;
  const int i   = blockIdx.x * blockDim.x + threadIdx.x;
  if (i >= tot) return;
  const int p = i % HW;
  const int c = (i / HW) % C;
  const int b = i / (HW * C);
  const int h = p / W, wp = p - (p / W) * W;
  const float px = (float)wp + flow[((long long)b * 2 + 0) * HW + p] * sx;
  const float py = (float)h  + flow[((long long)b * 2 + 1) * HW + p] * sy;
  const float x0 = floorf(px), y0 = floorf(py);
  const float tx = px - x0,   ty = py - y0;
  const float* img = x + ((long long)b * C + c) * HW;
  float val = 0.f, m = 0.f;
#pragma unroll
  for (int jy = 0; jy < 2; ++jy) {
#pragma unroll
    for (int jx = 0; jx < 2; ++jx) {
      const float ix = x0 + jx, iy = y0 + jy;
      const float wt = (jx ? tx : 1.f - tx) * (jy ? ty : 1.f - ty);
      if (ix >= 0.f && ix <= (float)(W - 1) && iy >= 0.f && iy <= (float)(H - 1)) {
        val += wt * img[(int)iy * W + (int)ix];
        m   += wt;
      }
    }
  }
  out[((long long)b * C + c) * HW + p] = (m >= 1.f - 1e-6f) ? val : 0.f;
}

// align-corners bilinear resize (matches separable _resize_ac), fused scale
__global__ void k_resize(const float* __restrict__ in, float* __restrict__ out,
                         int C, int Hi, int Wi, int Ho, int Wo, int B,
                         float ry, float rx, float scale)
{
  const int HWo = Ho * Wo;
  const int tot = B * C * HWo;
  const int i   = blockIdx.x * blockDim.x + threadIdx.x;
  if (i >= tot) return;
  const int p = i % HWo;
  const int c = (i / HWo) % C;
  const int b = i / (HWo * C);
  const int oh = p / Wo, ow = p - (p / Wo) * Wo;
  const float fy = oh * ry;
  const float fx = ow * rx;
  int y0 = (int)floorf(fy); int y1 = (y0 + 1 < Hi) ? y0 + 1 : Hi - 1;
  int x0 = (int)floorf(fx); int x1 = (x0 + 1 < Wi) ? x0 + 1 : Wi - 1;
  const float wy = fy - (float)y0, wx = fx - (float)x0;
  const float* s = in + ((long long)b * C + c) * (long long)Hi * Wi;
  const float v =
      (1.f - wy) * ((1.f - wx) * s[(long long)y0 * Wi + x0] + wx * s[(long long)y0 * Wi + x1]) +
      wy        * ((1.f - wx) * s[(long long)y1 * Wi + x0] + wx * s[(long long)y1 * Wi + x1]);
  out[((long long)b * C + c) * HWo + p] = v * scale;
}

__global__ void k_zero_ch(float* __restrict__ p, int Ctot, int coff, int nc, int HW, int B)
{
  const int tot = B * nc * HW;
  const int i   = blockIdx.x * blockDim.x + threadIdx.x;
  if (i >= tot) return;
  const int pix = i % HW;
  const int c   = (i / HW) % nc;
  const int b   = i / (HW * nc);
  p[((long long)b * Ctot + coff + c) * HW + pix] = 0.f;
}

__global__ void k_copy_ch(const float* __restrict__ s, int Cst, int cs,
                          float* __restrict__ d, int Cdt, int cd,
                          int nc, int HW, int B)
{
  const int tot = B * nc * HW;
  const int i   = blockIdx.x * blockDim.x + threadIdx.x;
  if (i >= tot) return;
  const int pix = i % HW;
  const int c   = (i / HW) % nc;
  const int b   = i / (HW * nc);
  d[((long long)b * Cdt + cd + c) * HW + pix] = s[((long long)b * Cst + cs + c) * HW + pix];
}

__global__ void k_add(float* __restrict__ a, const float* __restrict__ b, int n)
{
  const int i = blockIdx.x * blockDim.x + threadIdx.x;
  if (i < n) a[i] += b[i];
}

// ---------------------------------------------------------------------------
// Host orchestration
// ---------------------------------------------------------------------------
namespace {

struct ConvP { const float* w; const float* b; int cin, cout; };

inline int cdiv(int a, int b) { return (a + b - 1) / b; }

void launch_conv(hipStream_t st, int B,
                 const float* x, int CinTot, int cin_off, int Hin, int Win,
                 const ConvP& cp, float* y, int CoutTot, int cout_off,
                 int Hout, int Wout, int stride, int dil, int relu)
{
  const int M = Hout * Wout;
  dim3 g(cdiv(M, 32), cdiv(cp.cout, 64), B);
  k_conv3x3_wmma<<<g, 128, 0, st>>>(x, CinTot, cin_off, Hin, Win,
                                    cp.w, cp.b, y, CoutTot, cout_off,
                                    Hout, Wout, cp.cin, cp.cout, stride, dil, relu);
}

// dense flow-estimator block: outputs prepended into feat buffer (zero-copy
// concat), final 2-channel conv into flow_out. Input feat lives at chan 448.
void run_fe(hipStream_t st, int B, const ConvP* L, float* feat, int CtotF,
            int H, int W, float* flow_out)
{
  static const int offs[6] = {448, 320, 192, 96, 32, 0};
  for (int i = 0; i < 5; ++i)
    launch_conv(st, B, feat, CtotF, offs[i], H, W, L[i],
                feat, CtotF, offs[i + 1], H, W, 1, 1, 1);
  launch_conv(st, B, feat, CtotF, 0, H, W, L[5], flow_out, 2, 0, H, W, 1, 1, 0);
}

} // namespace

extern "C" void kernel_launch(void* const* d_in, const int* in_sizes, int n_in,
                              void* d_out, int out_size, void* d_ws, size_t ws_size,
                              hipStream_t stream)
{
  (void)in_sizes; (void)n_in; (void)out_size; (void)ws_size;
  const int   NB   = 2;
  const int   H_IM = 448, W_IM = 1024;
  const float DIVF = 0.05f;
  const int   TB   = 256;

  const float* x1 = (const float*)d_in[0];
  const float* x2 = (const float*)d_in[1];
  auto P = [&](int i) { return (const float*)d_in[i]; };

  // params pytree leaves, jax sorted-dict order: ctx, fe, fpe; conv dict: b, w
  static const int ctx_c[8] = {567, 128, 128, 128, 96, 64, 32, 2};
  ConvP ctx[7];
  for (int j = 0; j < 7; ++j) {
    ctx[j].b = P(2 + 2 * j); ctx[j].w = P(3 + 2 * j);
    ctx[j].cin = ctx_c[j]; ctx[j].cout = ctx_c[j + 1];
  }
  static const int ci_l[5] = {83, 213, 181, 149, 117};
  static const int feco[5] = {128, 128, 96, 64, 32};
  ConvP fe[5][6];
  for (int l = 0; l < 5; ++l) {
    int acc = ci_l[l];
    for (int i = 0; i < 6; ++i) {
      const int idx = 16 + (l * 6 + i) * 2;
      fe[l][i].b = P(idx); fe[l][i].w = P(idx + 1);
      if (i < 5) { fe[l][i].cin = acc; fe[l][i].cout = feco[i]; acc += feco[i]; }
      else       { fe[l][i].cin = acc; fe[l][i].cout = 2; }
    }
  }
  static const int chs[7] = {3, 16, 32, 64, 96, 128, 196};
  ConvP fpe[6][2];
  for (int k = 0; k < 6; ++k)
    for (int i = 0; i < 2; ++i) {
      const int idx = 76 + (k * 2 + i) * 2;
      fpe[k][i].b = P(idx); fpe[k][i].w = P(idx + 1);
      fpe[k][i].cin = (i == 0) ? chs[k] : chs[k + 1];
      fpe[k][i].cout = chs[k + 1];
    }

  // ---- workspace bump allocator (floats) ----------------------------------
  float* wsf = (float*)d_ws;
  size_t off = 0;
  auto alloc = [&](size_t n) { float* p = wsf + off; off += n; return p; };

  int pH[6], pW[6];
  float *p1[6], *p2[6];
  {
    int Hh = H_IM, Ww = W_IM;
    for (int k = 0; k < 6; ++k) {
      Hh = (Hh - 1) / 2 + 1; Ww = (Ww - 1) / 2 + 1;
      pH[k] = Hh; pW[k] = Ww;
      p1[k] = alloc((size_t)NB * chs[k + 1] * Hh * Ww);
      p2[k] = alloc((size_t)NB * chs[k + 1] * Hh * Ww);
    }
  }
  float* tmp = alloc((size_t)NB * 16 * pH[0] * pW[0]);
  const int HW4 = 112 * 256;
  float* feat = alloc((size_t)NB * 567 * HW4);   // widest dense-block buffer
  float* x2w  = alloc((size_t)NB * 32 * HW4);
  float* x1w  = alloc((size_t)NB * 32 * HW4);
  float* fa   = alloc((size_t)NB * 2 * HW4);
  float* fb   = alloc((size_t)NB * 2 * HW4);
  float* ga   = alloc((size_t)NB * 2 * HW4);
  float* gb   = alloc((size_t)NB * 2 * HW4);
  float* cflw = alloc((size_t)NB * 2 * HW4);
  float* cbA  = alloc((size_t)NB * 128 * HW4);
  float* cbB  = alloc((size_t)NB * 128 * HW4);

  // ---- feature pyramid (both images) --------------------------------------
  {
    const float* s1 = x1; const float* s2 = x2;
    int Hh = H_IM, Ww = W_IM;
    for (int k = 0; k < 6; ++k) {
      const int Ho = pH[k], Wo = pW[k], Co = chs[k + 1];
      launch_conv(stream, NB, s1, fpe[k][0].cin, 0, Hh, Ww, fpe[k][0], tmp, Co, 0, Ho, Wo, 2, 1, 1);
      launch_conv(stream, NB, tmp, Co, 0, Ho, Wo, fpe[k][1], p1[k], Co, 0, Ho, Wo, 1, 1, 1);
      launch_conv(stream, NB, s2, fpe[k][0].cin, 0, Hh, Ww, fpe[k][0], tmp, Co, 0, Ho, Wo, 2, 1, 1);
      launch_conv(stream, NB, tmp, Co, 0, Ho, Wo, fpe[k][1], p2[k], Co, 0, Ho, Wo, 1, 1, 1);
      s1 = p1[k]; s2 = p2[k]; Hh = Ho; Ww = Wo;
    }
  }

  // ---- coarse-to-fine levels ----------------------------------------------
  struct Lv { int C, H, W; };
  const Lv lv[5] = {{196, 7, 16}, {128, 14, 32}, {96, 28, 64}, {64, 56, 128}, {32, 112, 256}};
  float *fcur = fa, *falt = fb, *gcur = ga, *galt = gb;
  int pHh = 0, pWw = 0;

  for (int l = 0; l < 5; ++l) {
    const int C = lv[l].C, H = lv[l].H, W = lv[l].W, HW = H * W;
    const int ci = ci_l[l];
    const int CtotF = ci + 448 + (l == 4 ? 2 : 0);
    const float* f1 = p1[5 - l];
    const float* f2 = p2[5 - l];
    const float* xw2; const float* xw1;

    if (l > 0) {
      const float ry = (float)(pHh - 1) / (float)(H - 1);
      const float rx = (float)(pWw - 1) / (float)(W - 1);
      const int tot = NB * 2 * HW;
      k_resize<<<cdiv(tot, TB), TB, 0, stream>>>(fcur, falt, 2, pHh, pWw, H, W, NB, ry, rx, 1.f);
      { float* t = fcur; fcur = falt; falt = t; }
      k_resize<<<cdiv(tot, TB), TB, 0, stream>>>(gcur, galt, 2, pHh, pWw, H, W, NB, ry, rx, 1.f);
      { float* t = gcur; gcur = galt; galt = t; }
      const float sx = (float)(W - 1) / ((float)(W_IM - 1) * DIVF);
      const float sy = (float)(H - 1) / ((float)(H_IM - 1) * DIVF);
      const int tw = NB * C * HW;
      k_warp<<<cdiv(tw, TB), TB, 0, stream>>>(f2, fcur, x2w, C, H, W, NB, sx, sy);
      k_warp<<<cdiv(tw, TB), TB, 0, stream>>>(f1, gcur, x1w, C, H, W, NB, sx, sy);
      xw2 = x2w; xw1 = x1w;
    } else {
      xw2 = f2; xw1 = f1;
    }

    const int tcorr = NB * 81 * HW;
    const int tcopyC = NB * C * HW;
    const int tcopy2 = NB * 2 * HW;

    // ---- backward estimator (uses pre-update flows) -----------------------
    k_zero_ch<<<cdiv(tcopy2, TB), TB, 0, stream>>>(feat, CtotF, 448, 2, HW, NB);
    k_corr<<<cdiv(tcorr, TB), TB, 0, stream>>>(f2, xw1, feat, CtotF, 450, C, H, W, NB);
    if (l > 0) {
      k_copy_ch<<<cdiv(tcopyC, TB), TB, 0, stream>>>(f2, C, 0, feat, CtotF, 531, C, HW, NB);
      k_copy_ch<<<cdiv(tcopy2, TB), TB, 0, stream>>>(gcur, 2, 0, feat, CtotF, 531 + C, 2, HW, NB);
    }
    run_fe(stream, NB, fe[l], feat, CtotF, H, W, galt);
    { float* t = gcur; gcur = galt; galt = t; }

    // ---- forward estimator ------------------------------------------------
    k_zero_ch<<<cdiv(tcopy2, TB), TB, 0, stream>>>(feat, CtotF, 448, 2, HW, NB);
    k_corr<<<cdiv(tcorr, TB), TB, 0, stream>>>(f1, xw2, feat, CtotF, 450, C, H, W, NB);
    if (l > 0) {
      k_copy_ch<<<cdiv(tcopyC, TB), TB, 0, stream>>>(f1, C, 0, feat, CtotF, 531, C, HW, NB);
      k_copy_ch<<<cdiv(tcopy2, TB), TB, 0, stream>>>(fcur, 2, 0, feat, CtotF, 531 + C, 2, HW, NB);
    }
    run_fe(stream, NB, fe[l], feat, CtotF, H, W, falt);
    { float* t = fcur; fcur = falt; falt = t; }

    // ---- context net at finest level --------------------------------------
    if (l == 4) {
      k_copy_ch<<<cdiv(tcopy2, TB), TB, 0, stream>>>(fcur, 2, 0, feat, CtotF, 565, 2, HW, NB);
      const float* cin_p = feat;
      int cinTot = CtotF;
      float* outs[6] = {cbA, cbB, cbA, cbB, cbA, cbB};
      static const int dils[6] = {1, 2, 4, 8, 16, 1};
      for (int j = 0; j < 6; ++j) {
        launch_conv(stream, NB, cin_p, cinTot, 0, H, W, ctx[j],
                    outs[j], ctx[j].cout, 0, H, W, 1, dils[j], 1);
        cin_p = outs[j]; cinTot = ctx[j].cout;
      }
      launch_conv(stream, NB, cin_p, cinTot, 0, H, W, ctx[6], cflw, 2, 0, H, W, 1, 1, 0);
      k_add<<<cdiv(tcopy2, TB), TB, 0, stream>>>(fcur, cflw, tcopy2);
    }
    pHh = H; pWw = W;
  }

  // ---- final upsample to full res, scaled by 1/DIV_FLOW -------------------
  {
    const float ry = (float)(112 - 1) / (float)(H_IM - 1);
    const float rx = (float)(256 - 1) / (float)(W_IM - 1);
    const int tot = NB * 2 * H_IM * W_IM;
    k_resize<<<cdiv(tot, TB), TB, 0, stream>>>(fcur, (float*)d_out, 2, 112, 256,
                                               H_IM, W_IM, NB, ry, rx, 1.f / DIVF);
  }
}
